// RGCN_Attn_BlockLayer_63187558859233
// MI455X (gfx1250) — matrix operand order repaired and checked by reading, compiled, not run on hardware
//
#include <hip/hip_runtime.h>
#include <hip/hip_bf16.h>

// ---------------------------------------------------------------------------
// RGCN + attention layer for MI455X (gfx1250, wave32).
// Algebraic fusion: a2 = ev@Q is never materialized; instead
//   a[e] = dot(ev[e], b1[dst[e]])  with  b1 = h @ (K Q^T).
// Dense GEMMs use v_wmma_f32_16x16x32_bf16 (bf16 in / f32 accum), B fragments
// register-resident across 4 row tiles/wave, compile-time B strides so all
// fragment loads use immediate offsets, branch-free full-tile store epilogue.
// ev streams with NT hints to keep h / b1 / weight resident in the 192 MB L2.
// ---------------------------------------------------------------------------

typedef __attribute__((ext_vector_type(16))) __bf16 v16bf;
typedef __attribute__((ext_vector_type(8)))  float  v8f;

#define FEAT 128   // IN == OUT == 128
#define KDIM 128
#define RT_PER_WAVE 4

__device__ __forceinline__ void atomicMaxFloat(float* addr, float val) {
    // classic monotone-bit-pattern trick; m initialized to -inf bits
    if (val >= 0.0f) atomicMax((int*)addr, __float_as_int(val));
    else             atomicMin((unsigned int*)addr, __float_as_uint(val));
}

// ---------------------------------------------------------------------------
// C[M,128] = A[M,128] * B(128,128); B element (k,n) at Bp[k*BK + n*BN] with
// BK/BN compile-time -> all fragment loads are base + immediate offset.
// One wave owns one 16-wide column stripe; B fragments for all 4 K-steps are
// loaded once into registers, then 4 row tiles are swept (16 WMMAs / wave).
// ---------------------------------------------------------------------------
template <int BK, int BN>
__global__ void k_gemm_bf16(const float* __restrict__ A,
                            const float* __restrict__ Bp,
                            float* __restrict__ C, int Mrows) {
    const int nTiles = FEAT / 16;                       // 8 column tiles
    int wid  = (blockIdx.x * blockDim.x + threadIdx.x) >> 5;
    int lane = threadIdx.x & 31;
    int colTile  = wid % nTiles;
    int rowTile0 = (wid / nTiles) * RT_PER_WAVE;
    if (rowTile0 * 16 >= Mrows) return;                 // wave-uniform

    int nloc = lane & 15;                               // B column / D column
    int mloc = lane & 15;                               // A row within tile
    int kbA  = (lane >> 4) * 16;                        // A K-offset (contiguous, matches B)
    int kbB  = (lane >> 4) * 16;                        // B K-offset (contiguous, matches A)
    int colBase = colTile * 16;

    // B fragments for all 4 K-steps, resident for the whole wave lifetime.
    // Base pointer carries all runtime terms; loop offsets are constants.
    const float* bbase = Bp + (size_t)(colBase + nloc) * BN + (size_t)kbB * BK;
    v16bf fb[4];
    #pragma unroll
    for (int s = 0; s < 4; ++s) {
        #pragma unroll
        for (int q = 0; q < 8; ++q) {
            const int k = s * 32 + 2 * q;               // compile-time
            fb[s][2 * q]     = (__bf16)bbase[(k)     * BK];
            fb[s][2 * q + 1] = (__bf16)bbase[(k + 1) * BK];
        }
    }

    int mb = (lane >> 4) * 8;                           // D row offset (C/D layout)
    #pragma unroll
    for (int rt = 0; rt < RT_PER_WAVE; ++rt) {
        int rowBase = (rowTile0 + rt) * 16;
        if (rowBase >= Mrows) break;                    // wave-uniform
        int rowA = rowBase + mloc;
        if (rowA >= Mrows) rowA = Mrows - 1;            // clamp (stores are guarded)
        const float* abase = A + (size_t)rowA * FEAT + kbA;

        v8f acc = {};
        #pragma unroll
        for (int s = 0; s < 4; ++s) {
            v16bf fa;
            #pragma unroll
            for (int q = 0; q < 4; ++q) {
                const int k = s * 32 + 2 * q;           // compile-time
                fa[2 * q]     = (__bf16)abase[k];       // K = kb + {0..7}
                fa[2 * q + 1] = (__bf16)abase[k + 1];
                fa[2 * q + 8] = (__bf16)abase[k + 8];   // K = kb + {8..15}
                fa[2 * q + 9] = (__bf16)abase[k + 9];
            }
            acc = __builtin_amdgcn_wmma_f32_16x16x32_bf16(
                      false, fa, false, fb[s], (short)0, acc, false, false);
        }

        // C/D layout: VGPR q -> M = q + 8*(lane>=16); N = lane&15
        if (rowBase + 16 <= Mrows) {
            // full tile: branch-free, one base + 8 immediate-offset stores
            float* crow = C + (size_t)(rowBase + mb) * FEAT + colBase + nloc;
            #pragma unroll
            for (int q = 0; q < 8; ++q)
                crow[q * FEAT] = acc[q];
        } else {
            // generic tail tile (never taken for M % 16 == 0)
            #pragma unroll
            for (int q = 0; q < 8; ++q) {
                int row = rowBase + mb + q;
                if (row < Mrows)
                    C[(size_t)row * FEAT + colBase + nloc] = acc[q];
            }
        }
    }
}

// ---------------------------------------------------------------------------
// Init: out = 0, z = 0, m = -inf
// ---------------------------------------------------------------------------
__global__ void k_init(float* __restrict__ out, float* __restrict__ mbuf,
                       float* __restrict__ zbuf, int N) {
    int idx = blockIdx.x * blockDim.x + threadIdx.x;
    if (idx < N * FEAT) out[idx] = 0.0f;
    if (idx < N) {
        zbuf[idx] = 0.0f;
        mbuf[idx] = __uint_as_float(0xFF800000u);       // -inf
    }
}

// ---------------------------------------------------------------------------
// Per-edge (one wave32 per edge):
//   ev[e,:] = blockdiag(w[etype[e]]) applied to h[src[e]]   (B=8 blocks of 16x16)
//   a[e]    = dot(ev[e], b1[dst[e]]);  atomicMax(m[dst], a[e])
// h rows (25.6 MB) and w rows (512 KB) are L2-resident -> gathers hit cache.
// ev is streamed out with non-temporal stores (written once, read once).
// ---------------------------------------------------------------------------
__global__ void k_edge(const float* __restrict__ h, const float* __restrict__ weight,
                       const float* __restrict__ b1,
                       const int* __restrict__ src, const int* __restrict__ dst,
                       const int* __restrict__ etype,
                       float* __restrict__ ev, float* __restrict__ abuf,
                       float* __restrict__ mbuf, int E) {
    int wid  = (blockIdx.x * (blockDim.x >> 5)) + (threadIdx.x >> 5);
    int lane = threadIdx.x & 31;
    if (wid >= E) return;

    int s = src[wid], d = dst[wid], r = etype[wid];
    const float* hrow = h + (size_t)s * FEAT;
    const float* wr   = weight + (size_t)r * 2048;      // (B=8,16,16) row-major
    const float* brow = b1 + (size_t)d * FEAT;
    __builtin_prefetch(hrow + lane, 0, 0);              // global_prefetch_b8: warm h row

    float dotp = 0.0f;
    #pragma unroll
    for (int t = 0; t < 4; ++t) {
        int o  = lane + 32 * t;                         // output feature
        int b  = o >> 4;                                // block
        int oo = o & 15;
        const float* hb = hrow + b * 16;
        const float* wb = wr + b * 256 + oo;
        float acc = 0.0f;
        #pragma unroll
        for (int i = 0; i < 16; ++i) acc += hb[i] * wb[i * 16];
        __builtin_nontemporal_store(acc, ev + (size_t)wid * FEAT + o);  // stream, don't pollute L2
        dotp += acc * brow[o];
    }
    #pragma unroll
    for (int off = 16; off > 0; off >>= 1)
        dotp += __shfl_xor(dotp, off, 32);
    if (lane == 0) {
        abuf[wid] = dotp;
        atomicMaxFloat(mbuf + d, dotp);
    }
}

// ---------------------------------------------------------------------------
// s[e] = exp(a[e] - m[dst]);  z[dst] += s[e]
// ---------------------------------------------------------------------------
__global__ void k_expsum(const float* __restrict__ abuf, const int* __restrict__ dst,
                         const float* __restrict__ mbuf,
                         float* __restrict__ sbuf, float* __restrict__ zbuf, int E) {
    int e = blockIdx.x * blockDim.x + threadIdx.x;
    if (e >= E) return;
    int d = dst[e];
    float sv = __expf(abuf[e] - mbuf[d]);
    sbuf[e] = sv;
    atomicAdd(zbuf + d, sv);
}

// ---------------------------------------------------------------------------
// out[dst,:] += ev[e,:] * (s[e] / z[dst])   (one wave per edge, 4 feats/lane)
// ---------------------------------------------------------------------------
__global__ void k_scatter(const float* __restrict__ ev, const float* __restrict__ sbuf,
                          const float* __restrict__ zbuf, const int* __restrict__ dst,
                          float* __restrict__ out, int E) {
    int wid  = (blockIdx.x * (blockDim.x >> 5)) + (threadIdx.x >> 5);
    int lane = threadIdx.x & 31;
    if (wid >= E) return;
    const float* evrow = ev + (size_t)wid * FEAT;
    int d = dst[wid];
    float coeff = sbuf[wid] / zbuf[d];
    #pragma unroll
    for (int t = 0; t < 4; ++t) {
        int o = lane + 32 * t;
        float v = __builtin_nontemporal_load(evrow + o);   // streaming read of ev
        atomicAdd(out + (size_t)d * FEAT + o, v * coeff);
    }
}

// ---------------------------------------------------------------------------
extern "C" void kernel_launch(void* const* d_in, const int* in_sizes, int n_in,
                              void* d_out, int out_size, void* d_ws, size_t ws_size,
                              hipStream_t stream) {
    const float* h      = (const float*)d_in[0];
    const float* weight = (const float*)d_in[1];
    const float* attn_k = (const float*)d_in[2];   // [1,128,128] -> K[k,j] at k*128+j
    const float* attn_q = (const float*)d_in[3];   // [1,128,128] -> Q[i,j] at i*128+j
    const int*   src    = (const int*)d_in[4];
    const int*   dst    = (const int*)d_in[5];
    const int*   etype  = (const int*)d_in[6];
    float*       out    = (float*)d_out;

    const int N = in_sizes[0] / FEAT;
    const int E = in_sizes[4];

    // workspace carve-out
    char* ws = (char*)d_ws;
    size_t off = 0;
    auto carve = [&](size_t bytes) -> void* {
        void* p = ws + off;
        off += (bytes + 255) & ~(size_t)255;
        return p;
    };
    float* ev   = (float*)carve((size_t)E * FEAT * sizeof(float)); // 409.6 MB
    float* b1   = (float*)carve((size_t)N * FEAT * sizeof(float)); //  25.6 MB
    float* kq   = (float*)carve((size_t)KDIM * FEAT * sizeof(float));
    float* abuf = (float*)carve((size_t)E * sizeof(float));
    float* sbuf = (float*)carve((size_t)E * sizeof(float));
    float* mbuf = (float*)carve((size_t)N * sizeof(float));
    float* zbuf = (float*)carve((size_t)N * sizeof(float));
    (void)ws_size; (void)n_in; (void)out_size;

    // 0) init accumulators
    {
        int total = N * FEAT;
        k_init<<<(total + 255) / 256, 256, 0, stream>>>(out, mbuf, zbuf, N);
    }
    // 1) kq = K @ Q^T : kq[k,i] = sum_j K[k,j] * Q[i,j]
    //    B(k=j, n=i) at attn_q[i*128 + j] -> BK=1, BN=128
    {
        int rowTiles = KDIM / 16;                              // 8
        int waveRows = (rowTiles + RT_PER_WAVE - 1) / RT_PER_WAVE;
        int waves = waveRows * (FEAT / 16);                    // 16
        k_gemm_bf16<1, FEAT><<<(waves + 3) / 4, 128, 0, stream>>>(
            attn_k, attn_q, kq, KDIM);
    }
    // 2) b1 = h @ kq : B(k, n) at kq[k*128 + n] -> BK=128, BN=1
    {
        int rowTiles = (N + 15) / 16;                          // 3125 for N=50000
        int waveRows = (rowTiles + RT_PER_WAVE - 1) / RT_PER_WAVE;
        int waves = waveRows * (FEAT / 16);
        k_gemm_bf16<FEAT, 1><<<(waves + 3) / 4, 128, 0, stream>>>(
            h, kq, b1, N);
    }
    // 3) per-edge transform + score + segment max
    k_edge<<<(E + 7) / 8, 256, 0, stream>>>(h, weight, b1, src, dst, etype,
                                            ev, abuf, mbuf, E);
    // 4) softmax numerator + segment sum
    k_expsum<<<(E + 255) / 256, 256, 0, stream>>>(abuf, dst, mbuf, sbuf, zbuf, E);
    // 5) normalized scatter to destination nodes
    k_scatter<<<(E + 7) / 8, 256, 0, stream>>>(ev, sbuf, zbuf, dst, out, E);
}
